// GPSA_1382979470158
// MI455X (gfx1250) — matrix operand order repaired and checked by reading, not compiled, not run on hardware
//
#include <hip/hip_runtime.h>
#include <math.h>

typedef unsigned short ushort_t;
typedef __bf16 bf16;
typedef bf16  v16bf __attribute__((ext_vector_type(16)));
typedef float v8f   __attribute__((ext_vector_type(8)));

#define BB 16
#define NN 576
#define CC 768
#define HH 12
#define DD 64
#define GW 24

static __device__ __forceinline__ ushort_t f2bf(float f) {
  union { float f; unsigned u; } v; v.f = f;
  unsigned u = v.u;
  unsigned r = u + 0x7FFFu + ((u >> 16) & 1u);   // round-to-nearest-even
  return (ushort_t)(r >> 16);
}

union Frag16 { v16bf v; uint4 q[2]; ushort_t u[16]; };

static __device__ __forceinline__ v8f wmma_bf16(v16bf a, v16bf b, v8f c) {
  // D = A(16x32 bf16) x B(32x16 bf16) + C(16x16 f32)
  return __builtin_amdgcn_wmma_f32_16x16x32_bf16(false, a, false, b, (short)0, c,
                                                 false, false);
}

static __device__ __forceinline__ v8f vzero8() {
  v8f z;
#pragma unroll
  for (int i = 0; i < 8; i++) z[i] = 0.0f;
  return z;
}

// ---------------------------------------------------------------- converts
__global__ void cvt_f32_bf16(const float* __restrict__ src,
                             ushort_t* __restrict__ dst, int n) {
  int i = blockIdx.x * blockDim.x + threadIdx.x;
  if (i < n) dst[i] = f2bf(src[i]);
}

// Transpose a [K x Nc] f32 weight into col-major bf16 [Nc x K] (tiled, coalesced).
__global__ void __launch_bounds__(256)
transpose_w_bf16(const float* __restrict__ src, ushort_t* __restrict__ dst,
                 int rows, int cols) {  // rows=K, cols=Nc
  __shared__ float tile[32][33];
  int tx = threadIdx.x, ty = threadIdx.y;
  int r0 = blockIdx.y * 32, c0 = blockIdx.x * 32;
#pragma unroll
  for (int i = 0; i < 4; i++) {
    int r = r0 + ty + i * 8;
    tile[ty + i * 8][tx] = src[(size_t)r * cols + c0 + tx];
  }
  __syncthreads();
#pragma unroll
  for (int i = 0; i < 4; i++) {
    int n = c0 + ty + i * 8;
    int k = r0 + tx;
    dst[(size_t)n * rows + k] = f2bf(tile[tx][ty + i * 8]);
  }
}

// Transpose bf16 v [B,N,C(head-major)] -> vT [B,h,d,N] (tiled, coalesced).
__global__ void __launch_bounds__(256)
transpose_v_bf16(const ushort_t* __restrict__ vbf, ushort_t* __restrict__ vT) {
  __shared__ ushort_t tile[32][33];
  int tx = threadIdx.x, ty = threadIdx.y;
  int b = blockIdx.z / HH, h = blockIdx.z % HH;
  int n0 = blockIdx.x * 32, d0 = blockIdx.y * 32;
#pragma unroll
  for (int i = 0; i < 4; i++) {
    int n = n0 + ty + i * 8;
    tile[ty + i * 8][tx] = vbf[(size_t)(b * NN + n) * CC + h * DD + d0 + tx];
  }
  __syncthreads();
#pragma unroll
  for (int i = 0; i < 4; i++) {
    int d = d0 + ty + i * 8;
    vT[((size_t)(b * HH + h) * DD + d) * NN + n0 + tx] = tile[tx][ty + i * 8];
  }
}

// ------------------------------------------------- positional softmax [h,N,N]
__global__ void pos_softmax(const float* __restrict__ pos_w,
                            const float* __restrict__ pos_b,
                            ushort_t* __restrict__ posb) {
  __shared__ float red[256];
  int row = blockIdx.x;            // h*NN + n
  int h = row / NN, n = row % NN;
  int xn = n % GW, yn = n / GW;
  float w0 = pos_w[0 * HH + h], w1 = pos_w[1 * HH + h],
        w2 = pos_w[2 * HH + h], bb = pos_b[h];
  int t = threadIdx.x;

  float mx = -3.4e38f;
  for (int m = t; m < NN; m += 256) {
    int dx = (m % GW) - xn, dy = (m / GW) - yn;
    float val = (float)dx * w0 + (float)dy * w1 + (float)(dx * dx + dy * dy) * w2 + bb;
    mx = fmaxf(mx, val);
  }
  red[t] = mx; __syncthreads();
  for (int s = 128; s > 0; s >>= 1) {
    if (t < s) red[t] = fmaxf(red[t], red[t + s]);
    __syncthreads();
  }
  mx = red[0]; __syncthreads();

  float sm = 0.0f;
  for (int m = t; m < NN; m += 256) {
    int dx = (m % GW) - xn, dy = (m / GW) - yn;
    float val = (float)dx * w0 + (float)dy * w1 + (float)(dx * dx + dy * dy) * w2 + bb;
    sm += __expf(val - mx);
  }
  red[t] = sm; __syncthreads();
  for (int s = 128; s > 0; s >>= 1) {
    if (t < s) red[t] += red[t + s];
    __syncthreads();
  }
  sm = red[0]; __syncthreads();
  float inv = 1.0f / sm;

  for (int m = t; m < NN; m += 256) {
    int dx = (m % GW) - xn, dy = (m / GW) - yn;
    float val = (float)dx * w0 + (float)dy * w1 + (float)(dx * dx + dy * dy) * w2 + bb;
    posb[(size_t)row * NN + m] = f2bf(__expf(val - mx) * inv);
  }
}

// ------------------------------------------------- bf16 WMMA GEMM  C = A x Bt^T
// A [M x K] bf16 row-major, Bt [Nc x K] bf16 (pre-transposed weights).
// Tile: 128(M) x 128(N) x 32(K); 8 waves as 4(M) x 2(N), each wave 32x64.
__global__ void __launch_bounds__(256)
gemm_bf16(const ushort_t* __restrict__ A, const ushort_t* __restrict__ Bt,
          ushort_t* __restrict__ outBf, float* __restrict__ outF,
          const float* __restrict__ bias, int M, int Nc, int K) {
  __shared__ __align__(16) ushort_t ldsA[128 * 32];   // row-major [m][k]
  __shared__ __align__(16) ushort_t ldsB[128 * 32];   // col-major [n][k]
  int t = threadIdx.x;
  int lane = t & 31, wv = t >> 5;
  int half = lane >> 4, l16 = lane & 15;
  int wm = wv & 3, wn = wv >> 2;
  int mBase = blockIdx.y * 128, nBase = blockIdx.x * 128;

  v8f acc[2][4];
#pragma unroll
  for (int i = 0; i < 2; i++)
#pragma unroll
    for (int j = 0; j < 4; j++) acc[i][j] = vzero8();

  for (int kk = 0; kk < K; kk += 32) {
    __syncthreads();
    {  // A tile: 128 rows x 32 k, each thread 16 bf16 contiguous
      int row = t >> 1, kc = (t & 1) * 16;
      const uint4* g = reinterpret_cast<const uint4*>(
          A + (size_t)(mBase + row) * K + kk + kc);
      uint4* s = reinterpret_cast<uint4*>(&ldsA[row * 32 + kc]);
      s[0] = g[0]; s[1] = g[1];
    }
    {  // B tile: 128 n x 32 k, contiguous from pre-transposed weights
      int n = t >> 1, kc = (t & 1) * 16;
      const uint4* g = reinterpret_cast<const uint4*>(
          Bt + (size_t)(nBase + n) * K + kk + kc);
      uint4* s = reinterpret_cast<uint4*>(&ldsB[n * 32 + kc]);
      s[0] = g[0]; s[1] = g[1];
    }
    __syncthreads();

    Frag16 af[2], bfr[4];
#pragma unroll
    for (int mi = 0; mi < 2; mi++) {
      const ushort_t* p = &ldsA[(wm * 32 + mi * 16 + l16) * 32 + half * 8];
      af[mi].q[0] = *reinterpret_cast<const uint4*>(p);
      af[mi].q[1] = *reinterpret_cast<const uint4*>(p + 16);
    }
#pragma unroll
    for (int ni = 0; ni < 4; ni++) {
      const ushort_t* p = &ldsB[(wn * 64 + ni * 16 + l16) * 32 + half * 8];
      bfr[ni].q[0] = *reinterpret_cast<const uint4*>(p);
      bfr[ni].q[1] = *reinterpret_cast<const uint4*>(p + 16);
    }
#pragma unroll
    for (int mi = 0; mi < 2; mi++)
#pragma unroll
      for (int ni = 0; ni < 4; ni++)
        acc[mi][ni] = wmma_bf16(af[mi].v, bfr[ni].v, acc[mi][ni]);
  }

#pragma unroll
  for (int mi = 0; mi < 2; mi++)
#pragma unroll
    for (int ni = 0; ni < 4; ni++)
#pragma unroll
      for (int r = 0; r < 8; r++) {
        int row = mBase + wm * 32 + mi * 16 + half * 8 + r;
        int col = nBase + wn * 64 + ni * 16 + l16;
        float v = acc[mi][ni][r];
        if (outF)
          outF[(size_t)row * Nc + col] = v + (bias ? bias[col] : 0.0f);
        else
          outBf[(size_t)row * Nc + col] = f2bf(v);
      }
}

// ------------------------------------------------- fused gated attention
// grid (N/64, h, B); 128 threads = 4 waves; wave owns 16 query rows.
// out_head = (1-g) * softmax(q k^T * s) @ v  +  g * possm @ v
__global__ void __launch_bounds__(128)
gpsa_attn(const ushort_t* __restrict__ qb, const ushort_t* __restrict__ kb,
          const ushort_t* __restrict__ vT, const ushort_t* __restrict__ posb,
          const float* __restrict__ gating, ushort_t* __restrict__ attb) {
  __shared__ __align__(16) ushort_t qs[64 * 64];        // q block row-major
  __shared__ __align__(16) ushort_t ks[32 * 64];        // k tile row-major [m][d]
  __shared__ __align__(16) ushort_t vs[64 * 32];        // v tile col-major [d][m]
  __shared__ __align__(16) ushort_t pstage[4][16 * 32]; // per-wave exp(S) stage

  int t = threadIdx.x;
  int lane = t & 31, wv = t >> 5;
  int half = lane >> 4, l16 = lane & 15;
  int b = blockIdx.z, h = blockIdx.y, rowBase = blockIdx.x * 64;
  const float scale = 0.125f;  // d=64 -> 1/sqrt(64)
  size_t headCol = (size_t)h * DD;
  const ushort_t* vHead = vT + ((size_t)(b * HH + h) * DD) * NN;

  {  // load q block 64x64
    int row = t >> 1, ch = (t & 1) * 32;
    const uint4* g = reinterpret_cast<const uint4*>(
        qb + ((size_t)(b * NN + rowBase + row)) * CC + headCol + ch);
    uint4* s = reinterpret_cast<uint4*>(&qs[row * 64 + ch]);
    s[0] = g[0]; s[1] = g[1]; s[2] = g[2]; s[3] = g[3];
  }
  __syncthreads();

  Frag16 qa0, qa1;  // A-fragments of q: d 0..31 and d 32..63
  {
    const ushort_t* p = &qs[(wv * 16 + l16) * 64 + half * 8];
    qa0.q[0] = *reinterpret_cast<const uint4*>(p);
    qa0.q[1] = *reinterpret_cast<const uint4*>(p + 16);
    qa1.q[0] = *reinterpret_cast<const uint4*>(p + 32);
    qa1.q[1] = *reinterpret_cast<const uint4*>(p + 48);
  }

  v8f acc[4], pacc[4];
  float rs[8];
#pragma unroll
  for (int j = 0; j < 4; j++) { acc[j] = vzero8(); pacc[j] = vzero8(); }
#pragma unroll
  for (int r = 0; r < 8; r++) rs[r] = 0.0f;

  int qrow_g = rowBase + wv * 16 + l16;   // A-layout row for pos fragment
  const ushort_t* posRow = posb + ((size_t)h * NN + qrow_g) * NN;

  for (int mm0 = 0; mm0 < NN; mm0 += 32) {
    __syncthreads();
    {  // fill k tile (row-major [m][d]): 32 x 64
      int mr = t & 31, ch = (t >> 5) * 16;
      const uint4* gk = reinterpret_cast<const uint4*>(
          kb + ((size_t)(b * NN + mm0 + mr)) * CC + headCol + ch);
      uint4* sk = reinterpret_cast<uint4*>(&ks[mr * 64 + ch]);
      sk[0] = gk[0]; sk[1] = gk[1];
    }
    {  // fill v tile (col-major [d][m]) from pre-transposed vT: contiguous
      int d = t >> 1, mc = (t & 1) * 16;
      const uint4* gv = reinterpret_cast<const uint4*>(
          vHead + (size_t)d * NN + mm0 + mc);
      uint4* sv = reinterpret_cast<uint4*>(&vs[d * 32 + mc]);
      sv[0] = gv[0]; sv[1] = gv[1];
    }
    __syncthreads();

    // S = q k^T for two 16-col tiles, K=64 split as 2x32
    v8f s0 = vzero8(), s1 = vzero8();
#pragma unroll
    for (int nt = 0; nt < 2; nt++) {
      Frag16 bkl, bkh;
      const ushort_t* p0 = &ks[(nt * 16 + l16) * 64 + 0 + half * 8];
      const ushort_t* p1 = &ks[(nt * 16 + l16) * 64 + 32 + half * 8];
      bkl.q[0] = *reinterpret_cast<const uint4*>(p0);
      bkl.q[1] = *reinterpret_cast<const uint4*>(p0 + 16);
      bkh.q[0] = *reinterpret_cast<const uint4*>(p1);
      bkh.q[1] = *reinterpret_cast<const uint4*>(p1 + 16);
      if (nt == 0) {
        s0 = wmma_bf16(qa0.v, bkl.v, s0);
        s0 = wmma_bf16(qa1.v, bkh.v, s0);
      } else {
        s1 = wmma_bf16(qa0.v, bkl.v, s1);
        s1 = wmma_bf16(qa1.v, bkh.v, s1);
      }
    }

    // exp (scores are small; max-free is numerically safe), rowsum, stage bf16
    ushort_t* ps = &pstage[wv][0];
#pragma unroll
    for (int r = 0; r < 8; r++) {
      float e0 = __expf(s0[r] * scale);
      float e1 = __expf(s1[r] * scale);
      rs[r] += e0 + e1;
      int rr = half * 8 + r;
      ps[rr * 32 + l16]      = f2bf(e0);
      ps[rr * 32 + 16 + l16] = f2bf(e1);
    }
    __syncthreads();

    Frag16 pa;  // exp(S) as A fragment 16x32
    {
      const ushort_t* p = &pstage[wv][l16 * 32 + half * 8];
      pa.q[0] = *reinterpret_cast<const uint4*>(p);
      pa.q[1] = *reinterpret_cast<const uint4*>(p + 16);
    }
    Frag16 pg;  // positional softmax as A fragment 16x32 (from global)
    {
      const ushort_t* p = posRow + mm0 + half * 8;
      pg.q[0] = *reinterpret_cast<const uint4*>(p);
      pg.q[1] = *reinterpret_cast<const uint4*>(p + 16);
    }
#pragma unroll
    for (int j = 0; j < 4; j++) {
      Frag16 bv;
      const ushort_t* p = &vs[(j * 16 + l16) * 32 + half * 8];
      bv.q[0] = *reinterpret_cast<const uint4*>(p);
      bv.q[1] = *reinterpret_cast<const uint4*>(p + 16);
      acc[j]  = wmma_bf16(pa.v, bv.v, acc[j]);
      pacc[j] = wmma_bf16(pg.v, bv.v, pacc[j]);
    }
  }

  // rowsum reduce across the 16-lane group holding each row
#pragma unroll
  for (int r = 0; r < 8; r++) {
    float v = rs[r];
    v += __shfl_xor(v, 1, 32);
    v += __shfl_xor(v, 2, 32);
    v += __shfl_xor(v, 4, 32);
    v += __shfl_xor(v, 8, 32);
    rs[r] = v;
  }

  float g = 1.0f / (1.0f + __expf(-gating[h]));
#pragma unroll
  for (int r = 0; r < 8; r++) {
    int lr = half * 8 + r;
    int qr = rowBase + wv * 16 + lr;
    float inv = 1.0f / rs[r];
    size_t base = ((size_t)(b * NN + qr)) * CC + headCol;
#pragma unroll
    for (int j = 0; j < 4; j++) {
      float val = (1.0f - g) * acc[j][r] * inv + g * pacc[j][r];
      attb[base + j * 16 + l16] = f2bf(val);
    }
  }
}

// ----------------------------------------------------------------- launcher
extern "C" void kernel_launch(void* const* d_in, const int* in_sizes, int n_in,
                              void* d_out, int out_size, void* d_ws, size_t ws_size,
                              hipStream_t stream) {
  (void)in_sizes; (void)n_in; (void)out_size; (void)ws_size;
  const float* x      = (const float*)d_in[0];
  const float* Wq     = (const float*)d_in[1];
  const float* Wk     = (const float*)d_in[2];
  const float* Wv     = (const float*)d_in[3];
  const float* Wproj  = (const float*)d_in[4];
  const float* bproj  = (const float*)d_in[5];
  const float* pos_w  = (const float*)d_in[6];
  const float* pos_b  = (const float*)d_in[7];
  const float* gating = (const float*)d_in[8];
  float* out = (float*)d_out;

  char* w = (char*)d_ws;
  const size_t SX = (size_t)BB * NN * CC * 2;
  const size_t SW = (size_t)CC * CC * 2;
  const size_t SP = (size_t)HH * NN * NN * 2;
  ushort_t* xb  = (ushort_t*)w; w += SX;
  ushort_t* wqt = (ushort_t*)w; w += SW;   // col-major (pre-transposed)
  ushort_t* wkt = (ushort_t*)w; w += SW;
  ushort_t* wvt = (ushort_t*)w; w += SW;
  ushort_t* wpt = (ushort_t*)w; w += SW;
  ushort_t* qbf = (ushort_t*)w; w += SX;
  ushort_t* kbf = (ushort_t*)w; w += SX;
  ushort_t* vbf = (ushort_t*)w; w += SX;
  ushort_t* vTb = (ushort_t*)w; w += SX;   // [B,h,d,N]
  ushort_t* att = (ushort_t*)w; w += SX;
  ushort_t* pob = (ushort_t*)w; w += SP;

  const int nx = BB * NN * CC;
  cvt_f32_bf16<<<(nx + 255) / 256, 256, 0, stream>>>(x, xb, nx);

  dim3 tb(32, 8);
  dim3 tw(CC / 32, CC / 32);  // (24, 24)
  transpose_w_bf16<<<tw, tb, 0, stream>>>(Wq, wqt, CC, CC);
  transpose_w_bf16<<<tw, tb, 0, stream>>>(Wk, wkt, CC, CC);
  transpose_w_bf16<<<tw, tb, 0, stream>>>(Wv, wvt, CC, CC);
  transpose_w_bf16<<<tw, tb, 0, stream>>>(Wproj, wpt, CC, CC);

  pos_softmax<<<HH * NN, 256, 0, stream>>>(pos_w, pos_b, pob);

  dim3 gg(CC / 128, (BB * NN) / 128);  // (6, 72)
  gemm_bf16<<<gg, 256, 0, stream>>>(xb, wqt, qbf, nullptr, nullptr, BB * NN, CC, CC);
  gemm_bf16<<<gg, 256, 0, stream>>>(xb, wkt, kbf, nullptr, nullptr, BB * NN, CC, CC);
  gemm_bf16<<<gg, 256, 0, stream>>>(xb, wvt, vbf, nullptr, nullptr, BB * NN, CC, CC);

  transpose_v_bf16<<<dim3(NN / 32, DD / 32, BB * HH), tb, 0, stream>>>(vbf, vTb);

  gpsa_attn<<<dim3(NN / 64, HH, BB), 128, 0, stream>>>(qbf, kbf, vTb, pob, gating, att);

  gemm_bf16<<<gg, 256, 0, stream>>>(att, wpt, nullptr, out, bproj, BB * NN, CC, CC);
}
